// AttentionRNN_24137716203606
// MI455X (gfx1250) — matrix operand order repaired and checked by reading, not compiled, not run on hardware
//
#include <hip/hip_runtime.h>

// AttentionRNN forward for MI455X (gfx1250, wave32).
// V=32000, E=H=256, B=4, T=512. Dominant cost: fc GEMM (67 GFLOP) -> bf16 WMMA
// with 64x64 macro-tiles per wave to keep fc_W L2 traffic ~1 GB.
#define VV  32000
#define EE  256
#define HH  256
#define BBATCH 4
#define TT  512
#define MM  (BBATCH * TT)   // 2048
#define H3  (3 * HH)        // 768
#define K2H (2 * HH)        // 512

typedef __attribute__((ext_vector_type(2)))  float   v2f;
typedef __attribute__((ext_vector_type(8)))  float   v8f;
typedef __attribute__((ext_vector_type(4)))  __bf16  v4bf;
typedef __attribute__((ext_vector_type(8)))  __bf16  v8bf;
typedef __attribute__((ext_vector_type(16))) __bf16  v16bf;

// ---------------------------------------------------------------------------
// 1) Embedding gather: emb[m,:] = embed_W[x[m],:]
// ---------------------------------------------------------------------------
__global__ void gather_emb(const int* __restrict__ x, const float* __restrict__ embW,
                           float* __restrict__ emb) {
    const int m   = blockIdx.x;
    const int row = x[m];
    const float4* src = (const float4*)(embW + (size_t)row * EE);
    float4*       dst = (float4*)(emb + (size_t)m * EE);
    dst[threadIdx.x] = src[threadIdx.x];      // 64 threads x float4 = 256 floats
}

// ---------------------------------------------------------------------------
// 2) Generic f32 WMMA GEMM (K=256): C[m,n] = sum_k A[m,k]*Bt[n,k] + bias[n]
//    V_WMMA_F32_16X16X4_F32. Block = 128 thr (4 waves), wave = 16x16 tile.
//    A tile staged in LDS, pitch 260 floats -> bank-conflict-free col reads.
// ---------------------------------------------------------------------------
__global__ void gemm_wmma_f32(const float* __restrict__ A, const float* __restrict__ Bt,
                              const float* __restrict__ bias, float* __restrict__ C,
                              int N) {
    __shared__ float Ash[16 * 260];
    const int m0 = blockIdx.y * 16;
    for (int t = threadIdx.x; t < 16 * 64; t += blockDim.x) {  // 16 rows x 64 float4
        const int r = t >> 6, c4 = t & 63;
        float4 v = *(const float4*)(A + (size_t)(m0 + r) * EE + c4 * 4);
        *(float4*)(&Ash[r * 260 + c4 * 4]) = v;
    }
    __syncthreads();

    const int wave = threadIdx.x >> 5;
    const int lane = threadIdx.x & 31;
    const int hf   = lane >> 4;          // which half of the wave
    const int mr   = lane & 15;          // A row / B,C column role
    const int n    = blockIdx.x * 64 + wave * 16 + mr;
    const float* brow = Bt + (size_t)n * EE;

    v8f c = {};
    #pragma unroll
    for (int kb = 0; kb < 64; ++kb) {
        const int k0 = kb * 4 + hf * 2;                 // lanes<16: K=0,1  lanes>=16: K=2,3
        v2f a = *(const v2f*)(&Ash[mr * 260 + k0]);     // A[m][k0], A[m][k0+1]
        v2f b = *(const v2f*)(brow + k0);               // B[k0][n], B[k0+1][n] (Bt row-major)
        c = __builtin_amdgcn_wmma_f32_16x16x4_f32(false, a, false, b,
                                                  (short)0, c, false, false);
    }
    const float bv = bias[n];
    #pragma unroll
    for (int r = 0; r < 8; ++r) {                       // C: VGPR r -> M = r + 8*hf
        const int row = m0 + r + hf * 8;
        C[(size_t)row * N + n] = c[r] + bv;
    }
}

// ---------------------------------------------------------------------------
// 3) GRU scan: 512 sequential steps; single persistent workgroup (768 thr),
//    h state in LDS; each thread owns one of the 768 gate columns (all 4 batches).
// ---------------------------------------------------------------------------
__global__ void gru_scan(const float* __restrict__ xp, const float* __restrict__ h0,
                         const float* __restrict__ w_hh, const float* __restrict__ b_hh,
                         float* __restrict__ out, float* __restrict__ h_last) {
    __shared__ float hc[BBATCH * HH];   // 4x256 hidden state
    __shared__ float hp[BBATCH * H3];   // 4x768 gate pre-activations
    const int tid = threadIdx.x;        // 768
    for (int i = tid; i < BBATCH * HH; i += H3) hc[i] = h0[i];
    __syncthreads();

    const float4* wrow = (const float4*)(w_hh + (size_t)tid * HH);
    const float   bh   = b_hh[tid];

    for (int t = 0; t < TT; ++t) {
        float a0 = bh, a1 = bh, a2 = bh, a3 = bh;
        #pragma unroll 4
        for (int kk = 0; kk < HH / 4; ++kk) {
            const float4 w = wrow[kk];
            const int k = kk * 4;
            a0 += w.x*hc[0*HH+k] + w.y*hc[0*HH+k+1] + w.z*hc[0*HH+k+2] + w.w*hc[0*HH+k+3];
            a1 += w.x*hc[1*HH+k] + w.y*hc[1*HH+k+1] + w.z*hc[1*HH+k+2] + w.w*hc[1*HH+k+3];
            a2 += w.x*hc[2*HH+k] + w.y*hc[2*HH+k+1] + w.z*hc[2*HH+k+2] + w.w*hc[2*HH+k+3];
            a3 += w.x*hc[3*HH+k] + w.y*hc[3*HH+k+1] + w.z*hc[3*HH+k+2] + w.w*hc[3*HH+k+3];
        }
        hp[0*H3+tid] = a0; hp[1*H3+tid] = a1; hp[2*H3+tid] = a2; hp[3*H3+tid] = a3;
        __syncthreads();
        if (tid < HH) {
            #pragma unroll
            for (int b = 0; b < BBATCH; ++b) {
                const float* xpr = xp + (size_t)(b * TT + t) * H3;
                const float xr = xpr[tid], xz = xpr[HH + tid], xn = xpr[2*HH + tid];
                const float hr = hp[b*H3 + tid], hz = hp[b*H3 + HH + tid], hn = hp[b*H3 + 2*HH + tid];
                const float r  = 1.f / (1.f + __expf(-(xr + hr)));
                const float z  = 1.f / (1.f + __expf(-(xz + hz)));
                const float nn = tanhf(xn + r * hn);
                const float hnew = (1.f - z) * nn + z * hc[b*HH + tid];
                hc[b*HH + tid] = hnew;
                out[(size_t)(b * TT + t) * HH + tid] = hnew;
            }
        }
        __syncthreads();
    }
    for (int i = tid; i < BBATCH * HH; i += H3) h_last[i] = hc[i];
}

// ---------------------------------------------------------------------------
// 4) Fused additive attention per query row (b,i): tanh-score -> causal softmax
//    -> ctx; writes combined=[out|ctx] row directly as bf16 for the fc GEMM.
//    tanh is inside the H-reduction -> VALU/trans path, cannot be WMMA.
// ---------------------------------------------------------------------------
__global__ void attn_fused(const float* __restrict__ q, const float* __restrict__ kbuf,
                           const float* __restrict__ out, const float* __restrict__ vW,
                           const float* __restrict__ vb, __bf16* __restrict__ comb) {
    __shared__ float qrow[HH], vw[HH], sc[TT], red[256];
    const int tid = threadIdx.x;          // 256
    const int b   = blockIdx.x >> 9;
    const int i   = blockIdx.x & (TT - 1);
    const size_t m = (size_t)b * TT + i;

    qrow[tid] = q[m * HH + tid];
    vw[tid]   = vW[tid];
    __syncthreads();

    const int nj = i + 1;                 // causal: j in [0, i]
    for (int j = tid; j < nj; j += 256) {
        const float4* kr = (const float4*)(kbuf + ((size_t)b * TT + j) * HH);
        const float4* qr = (const float4*)qrow;
        const float4* vr = (const float4*)vw;
        float s = 0.f;
        #pragma unroll 4
        for (int h4 = 0; h4 < HH / 4; ++h4) {
            const float4 kv = kr[h4], qv = qr[h4], vv = vr[h4];
            s += vv.x * tanhf(qv.x + kv.x) + vv.y * tanhf(qv.y + kv.y)
               + vv.z * tanhf(qv.z + kv.z) + vv.w * tanhf(qv.w + kv.w);
        }
        sc[j] = s + vb[0];
    }
    __syncthreads();

    float lm = -3.0e38f;
    for (int j = tid; j < nj; j += 256) lm = fmaxf(lm, sc[j]);
    red[tid] = lm; __syncthreads();
    for (int s2 = 128; s2 > 0; s2 >>= 1) {
        if (tid < s2) red[tid] = fmaxf(red[tid], red[tid + s2]);
        __syncthreads();
    }
    const float mx = red[0]; __syncthreads();

    float ls = 0.f;
    for (int j = tid; j < nj; j += 256) { const float e = __expf(sc[j] - mx); sc[j] = e; ls += e; }
    red[tid] = ls; __syncthreads();
    for (int s2 = 128; s2 > 0; s2 >>= 1) {
        if (tid < s2) red[tid] += red[tid + s2];
        __syncthreads();
    }
    const float inv = 1.f / red[0];
    __syncthreads();

    float acc = 0.f;                      // ctx[h], h = tid; coalesced over out rows
    for (int j = 0; j < nj; ++j) acc += sc[j] * out[((size_t)b * TT + j) * HH + tid];
    acc *= inv;

    comb[m * K2H + tid]      = (__bf16)out[m * HH + tid];
    comb[m * K2H + HH + tid] = (__bf16)acc;
}

// ---------------------------------------------------------------------------
// 5) fc_W f32 -> bf16 (one-time; fc_W bf16 = 32 MB, fully L2-resident)
// ---------------------------------------------------------------------------
__global__ void cvt_bf16(const float* __restrict__ src, __bf16* __restrict__ dst, int n4) {
    const int i = blockIdx.x * blockDim.x + threadIdx.x;
    if (i < n4) {
        const float4 f = ((const float4*)src)[i];
        v4bf o = { (__bf16)f.x, (__bf16)f.y, (__bf16)f.z, (__bf16)f.w };
        ((v4bf*)dst)[i] = o;
    }
}

// ---------------------------------------------------------------------------
// 6) fc GEMM: logits[2048,32000] = comb[2048,512](bf16) @ fcW[32000,512](bf16)^T + b
//    V_WMMA_F32_16X16X32_BF16. Each wave computes a 64x64 macro-tile:
//    4 A-frags x 4 B-frags -> 16 WMMAs/k-step, 4x reuse of both operands.
//    Block = 128 thr (4 waves along N) -> block tile 64(M) x 256(N).
//    fc_W L2 traffic: (2048/64) passes x 32.8 MB = ~1.05 GB, balanced vs compute.
// ---------------------------------------------------------------------------
__global__ void fc_wmma_bf16(const __bf16* __restrict__ comb, const __bf16* __restrict__ fcw,
                             const float* __restrict__ fcb, float* __restrict__ logits) {
    const int wave  = threadIdx.x >> 5;     // 0..3
    const int lane  = threadIdx.x & 31;
    const int hf    = lane >> 4;
    const int lr    = lane & 15;
    const int m0    = blockIdx.y * 64;
    const int nbase = blockIdx.x * 256 + wave * 64;

    v8f acc[4][4];
    #pragma unroll
    for (int mt = 0; mt < 4; ++mt)
        #pragma unroll
        for (int nt = 0; nt < 4; ++nt)
            acc[mt][nt] = (v8f){};

    #pragma unroll
    for (int kb = 0; kb < K2H / 32; ++kb) {
        // A 16-bit layout: e<8 -> K=8*hf+e ; e>=8 -> K=16+8*hf+(e-8)
        v16bf a[4];
        #pragma unroll
        for (int mt = 0; mt < 4; ++mt) {
            const __bf16* ak = comb + (size_t)(m0 + mt * 16 + lr) * K2H + kb * 32;
            const v8bf alo = *(const v8bf*)(ak + 8 * hf);
            const v8bf ahi = *(const v8bf*)(ak + 16 + 8 * hf);
            a[mt] = __builtin_shufflevector(alo, ahi,
                        0,1,2,3,4,5,6,7,8,9,10,11,12,13,14,15);
        }
        const size_t koff = (size_t)kb * 32 + 8 * hf;   // B: e<8 -> K=8*hf+e ; e>=8 -> K=16+8*hf+(e-8), N = lr
        #pragma unroll
        for (int nt = 0; nt < 4; ++nt) {
            const __bf16* bk = fcw + (size_t)(nbase + nt * 16 + lr) * K2H + koff;
            const v16bf b = __builtin_shufflevector(*(const v8bf*)bk, *(const v8bf*)(bk + 16),
                        0,1,2,3,4,5,6,7,8,9,10,11,12,13,14,15);
            #pragma unroll
            for (int mt = 0; mt < 4; ++mt)
                acc[mt][nt] = __builtin_amdgcn_wmma_f32_16x16x32_bf16(
                                  false, a[mt], false, b, (short)0, acc[mt][nt], false, false);
        }
    }

    #pragma unroll
    for (int mt = 0; mt < 4; ++mt) {
        #pragma unroll
        for (int r = 0; r < 8; ++r) {                    // C: VGPR r -> M = r + 8*hf
            const int row = m0 + mt * 16 + r + hf * 8;
            float* lrow = logits + (size_t)row * VV;
            #pragma unroll
            for (int nt = 0; nt < 4; ++nt) {
                const int cn = nbase + nt * 16 + lr;
                lrow[cn] = acc[mt][nt][r] + fcb[cn];
            }
        }
    }
}

// ---------------------------------------------------------------------------
extern "C" void kernel_launch(void* const* d_in, const int* in_sizes, int n_in,
                              void* d_out, int out_size, void* d_ws, size_t ws_size,
                              hipStream_t stream) {
    (void)in_sizes; (void)n_in; (void)out_size; (void)ws_size;
    const int*   x    = (const int*)  d_in[0];
    const float* h0   = (const float*)d_in[1];
    const float* embW = (const float*)d_in[2];
    const float* w_ih = (const float*)d_in[3];
    const float* b_ih = (const float*)d_in[4];
    const float* w_hh = (const float*)d_in[5];
    const float* b_hh = (const float*)d_in[6];
    const float* awW  = (const float*)d_in[7];
    const float* awb  = (const float*)d_in[8];
    const float* auW  = (const float*)d_in[9];
    const float* aub  = (const float*)d_in[10];
    const float* avW  = (const float*)d_in[11];
    const float* avb  = (const float*)d_in[12];
    const float* fcW  = (const float*)d_in[13];
    const float* fcb  = (const float*)d_in[14];

    float* logits = (float*)d_out;
    float* h_last = logits + (size_t)MM * VV;   // outputs concatenated flat

    char* ws = (char*)d_ws;                     // ~47.3 MB total
    float*  emb  = (float*) (ws + ( 0ull << 20));   // 2 MB
    float*  xp   = (float*) (ws + ( 2ull << 20));   // 6 MB
    float*  gout = (float*) (ws + ( 8ull << 20));   // 2 MB
    float*  qb   = (float*) (ws + (10ull << 20));   // 2 MB
    float*  kb   = (float*) (ws + (12ull << 20));   // 2 MB
    __bf16* comb = (__bf16*)(ws + (14ull << 20));   // 2 MB
    __bf16* fcwb = (__bf16*)(ws + (16ull << 20));   // 31.25 MB

    gather_emb   <<<MM, 64, 0, stream>>>(x, embW, emb);
    gemm_wmma_f32<<<dim3(H3 / 64, MM / 16), 128, 0, stream>>>(emb, w_ih, b_ih, xp, H3);
    gru_scan     <<<1, H3, 0, stream>>>(xp, h0, w_hh, b_hh, gout, h_last);
    gemm_wmma_f32<<<dim3(HH / 64, MM / 16), 128, 0, stream>>>(gout, awW, awb, qb, HH);
    gemm_wmma_f32<<<dim3(HH / 64, MM / 16), 128, 0, stream>>>(gout, auW, aub, kb, HH);
    attn_fused   <<<MM, 256, 0, stream>>>(qb, kb, gout, avW, avb, comb);
    cvt_bf16     <<<(VV * K2H / 4 + 255) / 256, 256, 0, stream>>>(fcW, fcwb, VV * K2H / 4);
    fc_wmma_bf16 <<<dim3(VV / 256, MM / 64), 128, 0, stream>>>(comb, fcwb, fcb, logits);
}
